// Conv_83605833384007
// MI455X (gfx1250) — matrix operand order repaired and checked by reading, not compiled, not run on hardware
//
#include <hip/hip_runtime.h>

// ---------------------------------------------------------------------------
// Ensemble VALID 3x3 conv as implicit GEMM per ensemble member:
//   D[j][m] = sum_k W[j][k] * X[k][m],  M=C_out=128, N=62*62*32=123008, K=576
// bf16 WMMA (v_wmma_f32_16x16x32_bf16) with f32 accumulation.
// Block = 256 threads (8 wave32), tile = 128 (j) x 128 (pixels), K-step = 32.
// Full weight matrix for the ensemble member staged in LDS (CDNA5 320KB LDS).
// f32->bf16 now uses native converts (packed v_cvt) instead of bit-twiddling.
// ---------------------------------------------------------------------------

typedef __attribute__((ext_vector_type(16))) __bf16 v16bf;
typedef __attribute__((ext_vector_type(2)))  __bf16 v2bf;
typedef __attribute__((ext_vector_type(8)))  float  v8f;

struct U4x2 { uint4 lo, hi; };   // 32 bytes -> bit_cast to v16bf

// LDS layout (bytes)
#define WS_ROW_BF16   584u                  // 576 K + 8 pad (bank-conflict-free)
#define WS_BYTES      (128u * WS_ROW_BF16 * 2u)      // 149504
#define XS_ROW_BYTES  80u                   // 32 K bf16 + 8 pad
#define XS_BUF_BYTES  (128u * XS_ROW_BYTES)          // 10240
#define BIAS_OFF      (WS_BYTES + 2u * XS_BUF_BYTES) // 169984
#define SMEM_BYTES    (BIAS_OFF + 128u * 4u)         // 170496  (< 320KB)

__device__ __forceinline__ unsigned pack2_bf16(float a, float b) {
    v2bf p;
    p.x = (__bf16)a;            // RNE fptrunc -> hardware cvt (packed when possible)
    p.y = (__bf16)b;
    return __builtin_bit_cast(unsigned, p);
}

__global__ __launch_bounds__(256, 1)
void conv_ens_wmma_bf16(const float* __restrict__ feat,   // (4,32,64,64,64)
                        const float* __restrict__ wgt,    // (4,64,128,3,3)
                        const float* __restrict__ bias,   // (4,128)
                        float* __restrict__ out)          // (4,32,128,62,62)
{
    extern __shared__ char smem[];
    __bf16* Ws = (__bf16*)smem;                           // [128][584]
    char*   Xs = smem + WS_BYTES;                         // [2][128][80B]
    float*  Bs = (float*)(smem + BIAS_OFF);               // [128]

    const unsigned tid  = threadIdx.x;
    const unsigned lane = tid & 31u;
    const unsigned wave = tid >> 5;
    const unsigned half = lane >> 4;          // 0/1 (lane group)
    const unsigned l16  = lane & 15u;
    const unsigned waveJ = wave & 1u;         // 2 waves over j  (64 rows each)
    const unsigned waveP = wave >> 1;         // 4 waves over m  (32 pix each)

    const unsigned e     = blockIdx.x / 961u;
    const unsigned mt    = blockIdx.x - e * 961u;
    const unsigned mBase = mt * 128u;

    const float* fe = feat + (size_t)e * (32u * 64u * 64u * 64u);
    const float* we = wgt  + (size_t)e * (64u * 128u * 9u);
    const float* be = bias + (size_t)e * 128u;

    // ---- stage full weight matrix: Ws[j][kd*64 + c] = bf16(w[e][c][j][kd])
    for (unsigned l = tid; l < 73728u; l += 256u) {
        unsigned c   = l / 1152u;             // 1152 = 128*9
        unsigned rem = l - c * 1152u;
        unsigned j   = rem / 9u;
        unsigned kd  = rem - j * 9u;
        Ws[j * WS_ROW_BF16 + kd * 64u + c] = (__bf16)we[l];
    }
    if (tid < 128u) Bs[tid] = be[tid];

    // ---- per-thread pixel for X staging: 2 threads/pixel, 16 channels each
    const unsigned p    = tid & 127u;
    const unsigned cgrp = tid >> 7;           // 0/1 -> channel half of K-chunk
    const float* basePix;
    {
        unsigned m = mBase + p;
        unsigned b = m / 3844u;               // 3844 = 62*62
        unsigned r = m - b * 3844u;
        unsigned y = r / 62u;
        unsigned x = r - y * 62u;
        basePix = fe + (size_t)b * 262144u + y * 64u + x;   // channel 0, tap (0,0)
    }

    v8f acc[4][2] = {};                       // 4 j-tiles x 2 pixel-tiles

    // global loads for one K-step (step = kd*2 + ks; 32 channels, 16/thread)
    auto load_step = [&](unsigned step, float v[16]) {
        unsigned kd = step >> 1;
        unsigned dy = kd / 3u;
        unsigned dx = kd - dy * 3u;
        const float* src = basePix + dy * 64u + dx
                         + (size_t)((step & 1u) * 32u + cgrp * 16u) * 4096u;
#pragma unroll
        for (int j = 0; j < 16; ++j) v[j] = src[(size_t)j * 4096u];
    };

    // convert + store to LDS X buffer (2x ds_store_b128)
    auto store_step = [&](unsigned step, const float v[16]) {
        unsigned buf = step & 1u;
        unsigned pk[8];
#pragma unroll
        for (int j = 0; j < 8; ++j)
            pk[j] = pack2_bf16(v[2 * j], v[2 * j + 1]);
        uint4* dst = (uint4*)(Xs + buf * XS_BUF_BYTES + p * XS_ROW_BYTES + cgrp * 32u);
        dst[0] = make_uint4(pk[0], pk[1], pk[2], pk[3]);
        dst[1] = make_uint4(pk[4], pk[5], pk[6], pk[7]);
    };

    auto compute_step = [&](unsigned step) {
        unsigned buf   = step & 1u;
        unsigned koffB = ((step >> 1) * 64u + (step & 1u) * 32u) * 2u;

        // A fragments (weights): lane row M=l16; K = {half*8..+7, 16+half*8..+7}
        v16bf aF[4];
#pragma unroll
        for (int jt = 0; jt < 4; ++jt) {
            const char* rp = (const char*)Ws
                + (waveJ * 64u + jt * 16u + l16) * (WS_ROW_BF16 * 2u)
                + koffB + half * 16u;
            U4x2 t;
            t.lo = *(const uint4*)rp;
            t.hi = *(const uint4*)(rp + 32);
            aF[jt] = __builtin_bit_cast(v16bf, t);
        }
        // B fragments (pixels): lane col N=l16; K = {half*8..+7, 16+half*8..+7}
        v16bf bF[2];
#pragma unroll
        for (int pt = 0; pt < 2; ++pt) {
            const char* rp = Xs + buf * XS_BUF_BYTES
                + (waveP * 32u + pt * 16u + l16) * XS_ROW_BYTES + half * 16u;
            U4x2 t;
            t.lo = *(const uint4*)rp;
            t.hi = *(const uint4*)(rp + 32);
            bF[pt] = __builtin_bit_cast(v16bf, t);
        }
#pragma unroll
        for (int jt = 0; jt < 4; ++jt)
#pragma unroll
            for (int pt = 0; pt < 2; ++pt)
                acc[jt][pt] = __builtin_amdgcn_wmma_f32_16x16x32_bf16(
                    false, aF[jt], false, bF[pt], (short)0, acc[jt][pt],
                    false, false);
    };

    // ---- software-pipelined K loop: 9 taps x 2 K-chunks = 18 steps
    float v0[16], v1[16];
    load_step(0u, v0);
    store_step(0u, v0);
    __syncthreads();                          // weights + first X tile visible

    for (unsigned step = 0; step < 18u; ++step) {
        if (step + 1u < 18u) load_step(step + 1u, v1);   // overlap with compute
        compute_step(step);
        if (step + 1u < 18u) store_step(step + 1u, v1);  // other buffer
        __syncthreads();
    }

    // ---- epilogue: D lane layout => N (pixels) across lanes -> coalesced
#pragma unroll
    for (int pt = 0; pt < 2; ++pt) {
        unsigned m = mBase + waveP * 32u + pt * 16u + l16;
        unsigned b = m / 3844u;
        unsigned r = m - b * 3844u;
        unsigned y = r / 62u;
        unsigned x = r - y * 62u;
        float* ob = out + (size_t)(e * 32u + b) * 492032u   // 128*3844
                  + y * 62u + x;
#pragma unroll
        for (int jt = 0; jt < 4; ++jt) {
            unsigned j0 = waveJ * 64u + jt * 16u + half * 8u;
#pragma unroll
            for (int rr = 0; rr < 8; ++rr) {
                unsigned j = j0 + rr;
                ob[(size_t)j * 3844u] = acc[jt][pt][rr] + Bs[j];
            }
        }
    }
}

extern "C" void kernel_launch(void* const* d_in, const int* in_sizes, int n_in,
                              void* d_out, int out_size, void* d_ws, size_t ws_size,
                              hipStream_t stream) {
    (void)in_sizes; (void)n_in; (void)d_ws; (void)ws_size; (void)out_size;
    const float* feat = (const float*)d_in[0];
    const float* wgt  = (const float*)d_in[1];
    const float* bias = (const float*)d_in[2];
    float* out = (float*)d_out;

    // > 64KB dynamic LDS: opt in (CDNA5 allows up to 320KB per workgroup)
    hipFuncSetAttribute(reinterpret_cast<const void*>(conv_ens_wmma_bf16),
                        hipFuncAttributeMaxDynamicSharedMemorySize,
                        (int)SMEM_BYTES);

    // grid: 4 ensemble members x 961 pixel-tiles (123008 / 128, exact)
    conv_ens_wmma_bf16<<<4 * 961, 256, SMEM_BYTES, stream>>>(feat, wgt, bias, out);
}
